// DocumentClassificationGNN_47845935677470
// MI455X (gfx1250) — matrix operand order repaired and checked by reading, compiled, not run on hardware
//
#include <hip/hip_runtime.h>

// ---------------------------------------------------------------------------
// CDNA5 (gfx1250, wave32) GCN inference pipeline.
// GEMMs use v_wmma_f32_16x16x32_bf16 (fp32 accumulate).
// ---------------------------------------------------------------------------

typedef __attribute__((ext_vector_type(16))) __bf16 v16bf;
typedef __attribute__((ext_vector_type(8)))  float  v8f;

union BF16x16 { uint4 u[2]; v16bf v; };

__device__ __forceinline__ unsigned short f2bf(float f) {
    unsigned u = __float_as_uint(f);
    unsigned r = u + 0x7FFFu + ((u >> 16) & 1u);   // round-to-nearest-even
    return (unsigned short)(r >> 16);
}

// ---- degree / norm -------------------------------------------------------
__global__ void k_fill1(float* deg, int n) {
    int i = blockIdx.x * blockDim.x + threadIdx.x;
    if (i < n) deg[i] = 1.0f;                      // self-loop contributes 1
}
__global__ void k_degacc(float* deg, const long long* dst, int e) {
    int i = blockIdx.x * blockDim.x + threadIdx.x;
    if (i < e) atomicAdd(&deg[(int)dst[i]], 1.0f);
}
__global__ void k_rsqrt_ip(float* d, int n) {
    int i = blockIdx.x * blockDim.x + threadIdx.x;
    if (i < n) d[i] = rsqrtf(d[i]);
}

// ---- fp32 -> bf16 convert ------------------------------------------------
__global__ void k_cvt_bf16(const float* in, unsigned short* out, int n) {
    int i = blockIdx.x * blockDim.x + threadIdx.x;
    if (i < n) out[i] = f2bf(in[i]);
}

// ---- pack W[K][H] fp32 into WMMA B-fragment layout (bf16 pairs / dword) --
// dword d: i = d&7 (VGPR idx), lane = (d>>3)&31, tile t = d>>8,
//   nt = t % (H/16), kt = t / (H/16)
//   col = nt*16 + (lane&15); k0 = kt*32 + (lane>>4)*8 + (i>>2)*16 + 2*(i&3)
__global__ void k_packw(const float* W, unsigned int* out, int K, int H) {
    int d = blockIdx.x * blockDim.x + threadIdx.x;
    int total = (K * H) >> 1;
    if (d >= total) return;
    int i    = d & 7;
    int lane = (d >> 3) & 31;
    int t    = d >> 8;
    int NT   = H >> 4;
    int nt   = t % NT;
    int kt   = t / NT;
    int col  = nt * 16 + (lane & 15);
    int k0   = kt * 32 + (lane >> 4) * 8 + (i >> 2) * 16 + 2 * (i & 3);
    unsigned short lo = f2bf(W[(size_t)k0 * H + col]);
    unsigned short hi = f2bf(W[(size_t)(k0 + 1) * H + col]);
    out[d] = (unsigned)lo | ((unsigned)hi << 16);
}

// ---- WMMA GEMM: C[M x 128] = A_bf16[M x K] @ Wpacked ---------------------
// One wave per 16-row M-tile; 8 N-tiles (H=128) accumulated in v8f[8].
__global__ void __launch_bounds__(256) k_gemm_wmma(
        const unsigned short* __restrict__ A,
        const unsigned int*   __restrict__ Wp,
        float* __restrict__ C, int M, int K) {
    int wave  = threadIdx.x >> 5;
    int lane  = threadIdx.x & 31;
    int mtile = blockIdx.x * 8 + wave;
    if (mtile * 16 >= M) return;                   // wave-uniform: EXEC all-1s
    int khalf = lane >> 4;
    int l15   = lane & 15;
    int row   = mtile * 16 + l15;
    const unsigned short* arow = A + (size_t)row * K;

    v8f zero = {0.f,0.f,0.f,0.f,0.f,0.f,0.f,0.f};
    v8f acc[8];
#pragma unroll
    for (int nt = 0; nt < 8; ++nt) acc[nt] = zero;

    int ksteps = K >> 5;
    for (int kt = 0; kt < ksteps; ++kt) {
        int kb = kt * 32 + khalf * 8;
        BF16x16 a;
        a.u[0] = *(const uint4*)(arow + kb);        // K = kb..kb+7
        a.u[1] = *(const uint4*)(arow + kb + 16);   // K = kb+16..kb+23
        const uint4* bbase = (const uint4*)Wp + ((size_t)kt * 8) * 64 + lane * 2;
#pragma unroll
        for (int nt = 0; nt < 8; ++nt) {
            BF16x16 b;
            const uint4* bp = bbase + nt * 64;
            b.u[0] = bp[0];
            b.u[1] = bp[1];
            acc[nt] = __builtin_amdgcn_wmma_f32_16x16x32_bf16(
                false, a.v, false, b.v, (short)0, acc[nt], false, false);
        }
    }
    // C/D layout: VGPR i -> row (khalf*8 + i), col = l15
    int r0 = mtile * 16 + khalf * 8;
#pragma unroll
    for (int nt = 0; nt < 8; ++nt) {
        float* crow = C + (size_t)r0 * 128 + nt * 16 + l15;
#pragma unroll
        for (int i = 0; i < 8; ++i) crow[(size_t)i * 128] = acc[nt][i];
    }
}

// ---- aggregation init with self-loop: agg[i] = dinv[i]^2 * xw[i] ---------
__global__ void k_selfinit(float* agg, const float* xw, const float* dinv, int n) {
    int t = blockIdx.x * blockDim.x + threadIdx.x;
    if (t >= n * 32) return;
    int nd = t >> 5;
    int c  = (t & 31) * 4;
    float s = dinv[nd]; s = s * s;
    const float4 v = *(const float4*)(xw + (size_t)nd * 128 + c);
    float4 o; o.x = s * v.x; o.y = s * v.y; o.z = s * v.z; o.w = s * v.w;
    *(float4*)(agg + (size_t)nd * 128 + c) = o;
}

// ---- edge scatter: agg[dst] += dinv[src]*dinv[dst] * xw[src] -------------
__global__ void k_scatter(float* agg, const float* xw, const float* dinv,
                          const long long* src, const long long* dst, int e) {
    int t = blockIdx.x * blockDim.x + threadIdx.x;
    int ed = t >> 5;
    if (ed >= e) return;
    int c = (t & 31) * 4;
    int s = (int)src[ed];
    int d = (int)dst[ed];
    float nm = dinv[s] * dinv[d];
    const float4 v = *(const float4*)(xw + (size_t)s * 128 + c);
    float* o = agg + (size_t)d * 128 + c;
    atomicAdd(o + 0, nm * v.x);
    atomicAdd(o + 1, nm * v.y);
    atomicAdd(o + 2, nm * v.z);
    atomicAdd(o + 3, nm * v.w);
}

// ---- bias + BN + ReLU, emit bf16 for next GEMM ---------------------------
__global__ void k_bnrelu(const float* agg, const float* b,
                         const float* g, const float* beta,
                         const float* m, const float* v,
                         unsigned short* hbf, int total) {
    int i = blockIdx.x * blockDim.x + threadIdx.x;
    if (i >= total) return;
    int f = i & 127;
    float x = agg[i] + b[f];
    x = (x - m[f]) * rsqrtf(v[f] + 1e-5f) * g[f] + beta[f];
    x = fmaxf(x, 0.0f);
    hbf[i] = f2bf(x);
}

// ---- pooling -------------------------------------------------------------
__global__ void k_zero(float* p, int n) {
    int i = blockIdx.x * blockDim.x + threadIdx.x;
    if (i < n) p[i] = 0.0f;
}
__global__ void k_count(float* counts, const long long* batch, int n) {
    int i = blockIdx.x * blockDim.x + threadIdx.x;
    if (i < n) atomicAdd(&counts[(int)batch[i]], 1.0f);
}
__global__ void k_psum(float* sums, const float* agg, const long long* batch, int n) {
    int t = blockIdx.x * blockDim.x + threadIdx.x;
    int nd = t >> 5;
    if (nd >= n) return;
    int c = (t & 31) * 4;
    int g = (int)batch[nd];
    const float4 v = *(const float4*)(agg + (size_t)nd * 128 + c);
    float* o = sums + (size_t)g * 128 + c;
    atomicAdd(o + 0, v.x);
    atomicAdd(o + 1, v.y);
    atomicAdd(o + 2, v.z);
    atomicAdd(o + 3, v.w);
}

// ---- classifier MLP (64x128 -> 64 -> 16), single workgroup ---------------
__global__ void __launch_bounds__(256) k_classifier(
        const float* sums, const float* counts, const float* b3,
        const float* Wc1, const float* bc1,
        const float* Wc2, const float* bc2, float* out) {
    __shared__ float pooled[64 * 128];
    __shared__ float z[64 * 64];
    int t = threadIdx.x;
    for (int i = t; i < 64 * 128; i += 256) {
        int g = i >> 7, f = i & 127;
        float c = counts[g];
        pooled[i] = (c > 0.0f) ? (sums[i] / c + b3[f]) : 0.0f;
    }
    __syncthreads();
    for (int i = t; i < 64 * 64; i += 256) {
        int g = i >> 6, j = i & 63;
        float s = bc1[j];
        for (int h = 0; h < 128; ++h) s += pooled[g * 128 + h] * Wc1[h * 64 + j];
        z[i] = fmaxf(s, 0.0f);
    }
    __syncthreads();
    for (int i = t; i < 64 * 16; i += 256) {
        int g = i >> 4, c = i & 15;
        float s = bc2[c];
        for (int j = 0; j < 64; ++j) s += z[g * 64 + j] * Wc2[j * 16 + c];
        out[i] = s;
    }
}

// ---------------------------------------------------------------------------
static inline size_t alignup(size_t x) { return (x + 255) & ~(size_t)255; }

extern "C" void kernel_launch(void* const* d_in, const int* in_sizes, int n_in,
                              void* d_out, int out_size, void* d_ws, size_t ws_size,
                              hipStream_t stream) {
    const float*     x     = (const float*)d_in[0];
    const long long* ei    = (const long long*)d_in[1];
    const long long* batch = (const long long*)d_in[2];
    const float* W1 = (const float*)d_in[3];  const float* b1 = (const float*)d_in[4];
    const float* W2 = (const float*)d_in[5];  const float* b2 = (const float*)d_in[6];
    const float* W3 = (const float*)d_in[7];  const float* b3 = (const float*)d_in[8];
    const float* bn1g = (const float*)d_in[9];  const float* bn1b = (const float*)d_in[10];
    const float* bn1m = (const float*)d_in[11]; const float* bn1v = (const float*)d_in[12];
    const float* bn2g = (const float*)d_in[13]; const float* bn2b = (const float*)d_in[14];
    const float* bn2m = (const float*)d_in[15]; const float* bn2v = (const float*)d_in[16];
    const float* Wc1 = (const float*)d_in[17]; const float* bc1 = (const float*)d_in[18];
    const float* Wc2 = (const float*)d_in[19]; const float* bc2 = (const float*)d_in[20];
    float* out = (float*)d_out;

    const int N = in_sizes[0] / 256;
    const int E = in_sizes[1] / 2;
    const long long* src = ei;
    const long long* dst = ei + E;

    // workspace carve-up
    char* w = (char*)d_ws;
    float* dinv = (float*)w;                 w += alignup((size_t)N * 4);
    float* xw   = (float*)w;                 w += alignup((size_t)N * 128 * 4);
    float* agg  = (float*)w;                 w += alignup((size_t)N * 128 * 4);
    unsigned short* xbf = (unsigned short*)w; w += alignup((size_t)N * 256 * 2); // reused as hbf
    unsigned int* wp1 = (unsigned int*)w;    w += alignup((size_t)256 * 128 * 2);
    unsigned int* wp2 = (unsigned int*)w;    w += alignup((size_t)128 * 128 * 2);
    unsigned int* wp3 = (unsigned int*)w;    w += alignup((size_t)128 * 128 * 2);
    float* sums   = (float*)w;               w += alignup((size_t)64 * 128 * 4 + 64 * 4);
    float* counts = sums + 64 * 128;

    const int TB = 256;
    auto cdiv = [](int a, int b) { return (a + b - 1) / b; };

    // normalization: deg (with self-loops) -> dinv
    k_fill1<<<cdiv(N, TB), TB, 0, stream>>>(dinv, N);
    k_degacc<<<cdiv(E, TB), TB, 0, stream>>>(dinv, dst, E);
    k_rsqrt_ip<<<cdiv(N, TB), TB, 0, stream>>>(dinv, N);

    // convert inputs / pack weights into WMMA B-fragment layout
    k_cvt_bf16<<<cdiv(N * 256, TB), TB, 0, stream>>>(x, xbf, N * 256);
    k_packw<<<cdiv(256 * 128 / 2, TB), TB, 0, stream>>>(W1, wp1, 256, 128);
    k_packw<<<cdiv(128 * 128 / 2, TB), TB, 0, stream>>>(W2, wp2, 128, 128);
    k_packw<<<cdiv(128 * 128 / 2, TB), TB, 0, stream>>>(W3, wp3, 128, 128);

    const int mtiles = cdiv(N, 16);
    const int gemm_blocks = cdiv(mtiles, 8);

    // ---- layer 1 ----
    k_gemm_wmma<<<gemm_blocks, TB, 0, stream>>>(xbf, wp1, xw, N, 256);
    k_selfinit<<<cdiv(N * 32, TB), TB, 0, stream>>>(agg, xw, dinv, N);
    k_scatter<<<cdiv(E * 32, TB), TB, 0, stream>>>(agg, xw, dinv, src, dst, E);
    k_bnrelu<<<cdiv(N * 128, TB), TB, 0, stream>>>(agg, b1, bn1g, bn1b, bn1m, bn1v, xbf, N * 128);

    // ---- layer 2 ----
    k_gemm_wmma<<<gemm_blocks, TB, 0, stream>>>(xbf, wp2, xw, N, 128);
    k_selfinit<<<cdiv(N * 32, TB), TB, 0, stream>>>(agg, xw, dinv, N);
    k_scatter<<<cdiv(E * 32, TB), TB, 0, stream>>>(agg, xw, dinv, src, dst, E);
    k_bnrelu<<<cdiv(N * 128, TB), TB, 0, stream>>>(agg, b2, bn2g, bn2b, bn2m, bn2v, xbf, N * 128);

    // ---- layer 3 (no BN/ReLU; bias folded into pooling) ----
    k_gemm_wmma<<<gemm_blocks, TB, 0, stream>>>(xbf, wp3, xw, N, 128);
    k_selfinit<<<cdiv(N * 32, TB), TB, 0, stream>>>(agg, xw, dinv, N);
    k_scatter<<<cdiv(E * 32, TB), TB, 0, stream>>>(agg, xw, dinv, src, dst, E);

    // ---- global mean pool + classifier ----
    k_zero<<<cdiv(64 * 128 + 64, TB), TB, 0, stream>>>(sums, 64 * 128 + 64);
    k_count<<<cdiv(N, TB), TB, 0, stream>>>(counts, batch, N);
    k_psum<<<cdiv(N * 32, TB), TB, 0, stream>>>(sums, agg, batch, N);
    k_classifier<<<1, 256, 0, stream>>>(sums, counts, b3, Wc1, bc1, Wc2, bc2, out);
}
